// WindowAttention_81690277970126
// MI455X (gfx1250) — matrix operand order repaired and checked by reading, not compiled, not run on hardware
//
#include <hip/hip_runtime.h>

typedef _Float16 v16h __attribute__((ext_vector_type(16)));
typedef _Float16 v8h  __attribute__((ext_vector_type(8)));
typedef float    v8f  __attribute__((ext_vector_type(8)));
typedef int      v4i_ __attribute__((ext_vector_type(4)));

#define HEADS   12
#define NTOK    343
#define NPAD    352            // 22 tiles of 16
#define BATCH   128
#define DIM     384
#define M_TOT   (BATCH*NTOK)   // 43904 = 343*128
#define FQKV    1152

// ---- CDNA5 async global->LDS copy (ASYNCcnt-tracked), with safe fallback ----
// Probe round 2: clang-22 declares the builtin as (int4*, int4*, Imm, Imm) with
// generic pointers.
#if __has_builtin(__builtin_amdgcn_global_load_async_to_lds_b128)
#define ASYNC_B128(g, l)                                                      \
  __builtin_amdgcn_global_load_async_to_lds_b128((v4i_*)(g), (v4i_*)(l), 0, 0)
#else
#define ASYNC_B128(g, l) (*(v8h*)(l) = *(const v8h*)(g))
#endif

static __device__ __forceinline__ void wait_async_all() {
#if defined(__gfx1250__)
  asm volatile("s_wait_asynccnt 0x0" ::: "memory");
#endif
}

// ---------------------------------------------------------------- converts
__global__ __launch_bounds__(256) void cvt_f16_k(const float* __restrict__ in,
                                                 _Float16* __restrict__ out, int n) {
  for (int i = blockIdx.x * 256 + threadIdx.x; i < n; i += gridDim.x * 256)
    out[i] = (_Float16)in[i];
}

// bias_full[h][i][j] = bias_table[rel_index[i][j]*HEADS + h]  (L2-resident, reused 128x)
__global__ __launch_bounds__(256) void bias_expand_k(const float* __restrict__ tab,
                                                     const int* __restrict__ rel,
                                                     float* __restrict__ bf) {
  int ij = blockIdx.x * 256 + threadIdx.x;
  if (ij >= NTOK * NTOK) return;
  int idx = rel[ij];
#pragma unroll
  for (int h = 0; h < HEADS; ++h)
    bf[(size_t)h * NTOK * NTOK + ij] = tab[idx * HEADS + h];
}

// ---------------------------------------------------------------- QKV GEMM
// Y[m,f] = sum_k X[m,k] * Wqkv[f,k]; scatter to q/k/v in [B,H,N,32] f16, q pre-scaled.
// Block 128(M) x 64(F), 8 waves, double-buffered async LDS pipeline over K.
__global__ __launch_bounds__(256) void qkv_gemm_k(const _Float16* __restrict__ A,
                                                  const _Float16* __restrict__ W,
                                                  _Float16* __restrict__ qo,
                                                  _Float16* __restrict__ ko,
                                                  _Float16* __restrict__ vo,
                                                  float scale) {
  __shared__ __align__(64) _Float16 lA[2][128 * 32];
  __shared__ __align__(64) _Float16 lW[2][64 * 32];
  const int mBlk = blockIdx.x * 128;
  const int fBlk = blockIdx.y * 64;
  const int tid = threadIdx.x, wave = tid >> 5, lane = tid & 31;
  const int mW = (wave >> 2) * 64;        // 0 / 64
  const int fW = (wave & 3) * 16;         // 0..48
  v8f acc[4] = {v8f{}, v8f{}, v8f{}, v8f{}};

  // per-thread staging slices
  const int rA = tid >> 1, cA = (tid & 1) * 16;   // A: 16 halves (2 x b128)
  const int rW = tid >> 2, cW = (tid & 3) * 8;    // W: 8 halves  (1 x b128)
  const _Float16* gA = &A[(size_t)(mBlk + rA) * DIM + cA];
  const _Float16* gW = &W[(size_t)(fBlk + rW) * DIM + cW];

  auto stage = [&](int k0, int buf) {
    _Float16* la = &lA[buf][rA * 32 + cA];
    ASYNC_B128(gA + k0, la);
    ASYNC_B128(gA + k0 + 8, la + 8);
    ASYNC_B128(gW + k0, &lW[buf][rW * 32 + cW]);
  };

  stage(0, 0);
  wait_async_all();
  __syncthreads();

  int buf = 0;
#pragma unroll
  for (int step = 0; step < DIM / 32; ++step) {
    if (step + 1 < DIM / 32) stage((step + 1) * 32, buf ^ 1);  // overlap with WMMAs
    // B fragment: lane<16 -> col n=lane, K 0..15 ; lane>=16 -> col lane-16, K 16..31
    v16h bfr = *(const v16h*)&lW[buf][(fW + (lane & 15)) * 32 + ((lane < 16) ? 0 : 16)];
#pragma unroll
    for (int s = 0; s < 4; ++s) {
      int arow = mW + s * 16 + (lane & 15);
      int koff = (lane < 16) ? 0 : 16;  // A: K 0..15 | 16..31
      v8h a0 = *(const v8h*)&lA[buf][arow * 32 + koff];
      v8h a1 = *(const v8h*)&lA[buf][arow * 32 + koff + 8];
      v16h af = __builtin_shufflevector(a0, a1, 0,1,2,3,4,5,6,7,8,9,10,11,12,13,14,15);
      acc[s] = __builtin_amdgcn_wmma_f32_16x16x32_f16(false, af, false, bfr,
                                                      (short)0, acc[s], false, false);
    }
    wait_async_all();
    __syncthreads();
    buf ^= 1;
  }

  // scatter: fcol -> (which, head, dd)
  int fcol = fBlk + fW + (lane & 15);
  int which = fcol / DIM, inner = fcol % DIM;
  int h = inner >> 5, dd = inner & 31;
  _Float16* dst = (which == 0) ? qo : (which == 1) ? ko : vo;
  float mul = (which == 0) ? scale : 1.0f;
#pragma unroll
  for (int s = 0; s < 4; ++s)
#pragma unroll
    for (int r = 0; r < 8; ++r) {
      int m = mBlk + mW + s * 16 + r + ((lane >> 4) << 3);
      int b = m / NTOK, n = m % NTOK;
      dst[((((size_t)b * HEADS + h) * NTOK + n) << 5) + dd] = (_Float16)(acc[s][r] * mul);
    }
}

// ---------------------------------------------------------------- fused attention
// One block per (b,h). 8 waves; wave owns q row-tiles {w, w+8, w+16}.
// Flash-style online softmax over 11 key-tile pairs (K=32 per PV step).
__global__ __launch_bounds__(256) void attn_k(const _Float16* __restrict__ qg,
                                              const _Float16* __restrict__ kg,
                                              const _Float16* __restrict__ vg,
                                              const float* __restrict__ biasf,
                                              const float* __restrict__ mask,
                                              _Float16* __restrict__ ao) {
  __shared__ __align__(64) _Float16 ks[NPAD * 32];       // K rows, row-major
  __shared__ __align__(64) _Float16 vt[32 * NPAD];       // V transposed [d][NPAD]
  __shared__ __align__(64) _Float16 pbuf[8 * 16 * 32];   // per-wave P re-layout buffer
  const int b = blockIdx.x, h = blockIdx.y;
  const int tid = threadIdx.x, wave = tid >> 5, lane = tid & 31;
  const size_t base = (((size_t)b * HEADS + h) * NTOK) << 5;

  // zero vt and the ks padding tail; stream the ks body async into LDS
  for (int i = tid; i < 32 * NPAD; i += 256) vt[i] = (_Float16)0.f;
  for (int i = NTOK * 32 + tid; i < NPAD * 32; i += 256) ks[i] = (_Float16)0.f;
  for (int c = tid; c < (NTOK * 32) / 8; c += 256)
    ASYNC_B128(kg + base + (size_t)c * 8, ks + c * 8);
  __syncthreads();                       // vt zeros visible before transpose fill
  for (int i = tid; i < NTOK * 32; i += 256) {
    int row = i >> 5, c = i & 31;
    vt[c * NPAD + row] = vg[base + i];
  }
  wait_async_all();
  __syncthreads();

  const float NEG = -1e30f;
  const float* brow = biasf + (size_t)h * NTOK * NTOK;
  const float* mrow = mask + (size_t)(b & 31) * NTOK * NTOK;
  _Float16* pb = &pbuf[wave * 16 * 32];
  const int jcol = lane & 15;

  for (int qt = wave; qt < NPAD / 16; qt += 8) {
    const int qBase = qt * 16;
    // q A-fragment straight from global (rows clamped; garbage rows never stored)
    int qrow = qBase + (lane & 15); if (qrow > NTOK - 1) qrow = NTOK - 1;
    const _Float16* qp = qg + base + ((size_t)qrow << 5);
    int koff = (lane < 16) ? 0 : 16;
    v8h q0 = *(const v8h*)(qp + koff);
    v8h q1 = *(const v8h*)(qp + koff + 8);
    v16h qf = __builtin_shufflevector(q0, q1, 0,1,2,3,4,5,6,7,8,9,10,11,12,13,14,15);

    float mrw[8], lrw[8];
#pragma unroll
    for (int r = 0; r < 8; ++r) { mrw[r] = NEG; lrw[r] = 0.f; }
    v8f o0 = v8f{}, o1 = v8f{};
    const int iBase = qBase + ((lane >> 4) << 3);

    for (int jp = 0; jp < NPAD / 32; ++jp) {
      const int kBase = jp * 32;
      int dd = (lane < 16) ? 0 : 16;
      v16h kb0 = *(const v16h*)&ks[(kBase + jcol) * 32 + dd];
      v16h kb1 = *(const v16h*)&ks[(kBase + 16 + jcol) * 32 + dd];
      v8f s0 = v8f{}, s1 = v8f{};
      s0 = __builtin_amdgcn_wmma_f32_16x16x32_f16(false, qf, false, kb0, (short)0, s0, false, false);
      s1 = __builtin_amdgcn_wmma_f32_16x16x32_f16(false, qf, false, kb1, (short)0, s1, false, false);

#pragma unroll
      for (int r = 0; r < 8; ++r) {
        int i = iBase + r; int ii = (i < NTOK) ? i : NTOK - 1;
        int j0 = kBase + jcol, j1 = j0 + 16;
        float t0 = s0[r], t1 = s1[r];
        if (j0 < NTOK) t0 += brow[(size_t)ii * NTOK + j0] + mrow[(size_t)ii * NTOK + j0]; else t0 = NEG;
        if (j1 < NTOK) t1 += brow[(size_t)ii * NTOK + j1] + mrow[(size_t)ii * NTOK + j1]; else t1 = NEG;
        float vmax = fmaxf(t0, t1);
#pragma unroll
        for (int mk = 8; mk >= 1; mk >>= 1) vmax = fmaxf(vmax, __shfl_xor(vmax, mk, 32));
        float newm = fmaxf(mrw[r], vmax);
        float alpha = __expf(mrw[r] - newm);
        float p0 = __expf(t0 - newm), p1 = __expf(t1 - newm);
        float psum = p0 + p1;
#pragma unroll
        for (int mk = 8; mk >= 1; mk >>= 1) psum += __shfl_xor(psum, mk, 32);
        lrw[r] = lrw[r] * alpha + psum;
        mrw[r] = newm;
        o0[r] *= alpha; o1[r] *= alpha;
        int prow = r + ((lane >> 4) << 3);
        pb[prow * 32 + jcol]      = (_Float16)p0;   // D layout -> [row][K] staging
        pb[prow * 32 + 16 + jcol] = (_Float16)p1;
      }
      // P as A-fragment (same-wave LDS is in-order: no barrier needed)
      {
        int pr = lane & 15, pk = (lane < 16) ? 0 : 16;
        v8h p0v = *(const v8h*)&pb[pr * 32 + pk];
        v8h p1v = *(const v8h*)&pb[pr * 32 + pk + 8];
        v16h pf = __builtin_shufflevector(p0v, p1v, 0,1,2,3,4,5,6,7,8,9,10,11,12,13,14,15);
        v16h vb0 = *(const v16h*)&vt[jcol * NPAD + kBase + dd];
        v16h vb1 = *(const v16h*)&vt[(16 + jcol) * NPAD + kBase + dd];
        o0 = __builtin_amdgcn_wmma_f32_16x16x32_f16(false, pf, false, vb0, (short)0, o0, false, false);
        o1 = __builtin_amdgcn_wmma_f32_16x16x32_f16(false, pf, false, vb1, (short)0, o1, false, false);
      }
    }
#pragma unroll
    for (int r = 0; r < 8; ++r) {
      int i = iBase + r;
      if (i >= NTOK) continue;
      float inv = (lrw[r] > 0.f) ? 1.f / lrw[r] : 0.f;
      size_t o = ((size_t)(b * NTOK + i)) * DIM + h * 32;   // [B,N,C] layout
      ao[o + jcol]      = (_Float16)(o0[r] * inv);
      ao[o + 16 + jcol] = (_Float16)(o1[r] * inv);
    }
  }
}

// ---------------------------------------------------------------- projection GEMM
__global__ __launch_bounds__(256) void proj_gemm_k(const _Float16* __restrict__ A,
                                                   const _Float16* __restrict__ W,
                                                   const float* __restrict__ bias,
                                                   float* __restrict__ out) {
  __shared__ __align__(64) _Float16 lA[2][128 * 32];
  __shared__ __align__(64) _Float16 lW[2][64 * 32];
  const int mBlk = blockIdx.x * 128;
  const int fBlk = blockIdx.y * 64;
  const int tid = threadIdx.x, wave = tid >> 5, lane = tid & 31;
  const int mW = (wave >> 2) * 64;
  const int fW = (wave & 3) * 16;
  v8f acc[4] = {v8f{}, v8f{}, v8f{}, v8f{}};

  const int rA = tid >> 1, cA = (tid & 1) * 16;
  const int rW = tid >> 2, cW = (tid & 3) * 8;
  const _Float16* gA = &A[(size_t)(mBlk + rA) * DIM + cA];
  const _Float16* gW = &W[(size_t)(fBlk + rW) * DIM + cW];

  auto stage = [&](int k0, int buf) {
    _Float16* la = &lA[buf][rA * 32 + cA];
    ASYNC_B128(gA + k0, la);
    ASYNC_B128(gA + k0 + 8, la + 8);
    ASYNC_B128(gW + k0, &lW[buf][rW * 32 + cW]);
  };

  stage(0, 0);
  wait_async_all();
  __syncthreads();

  int buf = 0;
#pragma unroll
  for (int step = 0; step < DIM / 32; ++step) {
    if (step + 1 < DIM / 32) stage((step + 1) * 32, buf ^ 1);
    v16h bfr = *(const v16h*)&lW[buf][(fW + (lane & 15)) * 32 + ((lane < 16) ? 0 : 16)];
#pragma unroll
    for (int s = 0; s < 4; ++s) {
      int arow = mW + s * 16 + (lane & 15);
      int koff = (lane < 16) ? 0 : 16;
      v8h a0 = *(const v8h*)&lA[buf][arow * 32 + koff];
      v8h a1 = *(const v8h*)&lA[buf][arow * 32 + koff + 8];
      v16h af = __builtin_shufflevector(a0, a1, 0,1,2,3,4,5,6,7,8,9,10,11,12,13,14,15);
      acc[s] = __builtin_amdgcn_wmma_f32_16x16x32_f16(false, af, false, bfr,
                                                      (short)0, acc[s], false, false);
    }
    wait_async_all();
    __syncthreads();
    buf ^= 1;
  }

  int fcol = fBlk + fW + (lane & 15);
  float bv = bias[fcol];
#pragma unroll
  for (int s = 0; s < 4; ++s)
#pragma unroll
    for (int r = 0; r < 8; ++r) {
      int m = mBlk + mW + s * 16 + r + ((lane >> 4) << 3);
      out[(size_t)m * DIM + fcol] = acc[s][r] + bv;
    }
}

// ---------------------------------------------------------------- launch
extern "C" void kernel_launch(void* const* d_in, const int* in_sizes, int n_in,
                              void* d_out, int out_size, void* d_ws, size_t ws_size,
                              hipStream_t stream) {
  (void)in_sizes; (void)n_in; (void)out_size; (void)ws_size;
  const float* x          = (const float*)d_in[0];
  const float* mask       = (const float*)d_in[4];
  const float* w_qkv      = (const float*)d_in[6];
  const float* bias_table = (const float*)d_in[7];
  const float* w_proj     = (const float*)d_in[8];
  const float* b_proj     = (const float*)d_in[9];
  const int*   rel_index  = (const int*)d_in[10];
  float* out = (float*)d_out;

  char* ws = (char*)d_ws;
  size_t off = 0;
  auto carve = [&](size_t bytes) {
    void* p = ws + off;
    off = (off + bytes + 255) & ~(size_t)255;
    return p;
  };
  const size_t nX = (size_t)M_TOT * DIM;            // 16,859,136
  const size_t nQ = (size_t)BATCH * HEADS * NTOK * 32;
  _Float16* xf16 = (_Float16*)carve(nX * 2);
  _Float16* wq16 = (_Float16*)carve((size_t)FQKV * DIM * 2);
  _Float16* wp16 = (_Float16*)carve((size_t)DIM * DIM * 2);
  _Float16* q16  = (_Float16*)carve(nQ * 2);
  _Float16* k16  = (_Float16*)carve(nQ * 2);
  _Float16* v16  = (_Float16*)carve(nQ * 2);
  _Float16* ao16 = (_Float16*)carve(nX * 2);
  float*    bfull = (float*)carve((size_t)HEADS * NTOK * NTOK * 4);

  // 1) converts
  cvt_f16_k<<<4096, 256, 0, stream>>>(x, xf16, (int)nX);
  cvt_f16_k<<<432, 256, 0, stream>>>(w_qkv, wq16, FQKV * DIM);
  cvt_f16_k<<<144, 256, 0, stream>>>(w_proj, wp16, DIM * DIM);
  // 2) bias expand
  bias_expand_k<<<(NTOK * NTOK + 255) / 256, 256, 0, stream>>>(bias_table, rel_index, bfull);
  // 3) QKV GEMM  (343 M-blocks x 18 F-blocks)
  qkv_gemm_k<<<dim3(M_TOT / 128, FQKV / 64), 256, 0, stream>>>(
      xf16, wq16, q16, k16, v16, 0.17677669529663687f /* 32^-0.5 */);
  // 4) fused attention, one block per (b,h)
  attn_k<<<dim3(BATCH, HEADS), 256, 0, stream>>>(q16, k16, v16, bfull, mask, ao16);
  // 5) projection GEMM (343 x 6)
  proj_gemm_k<<<dim3(M_TOT / 128, DIM / 64), 256, 0, stream>>>(ao16, wp16, b_proj, out);
}
